// SwinTransformerBlock_1451698946257
// MI455X (gfx1250) — matrix operand order repaired and checked by reading, not compiled, not run on hardware
//
#include <hip/hip_runtime.h>
#include <hip/hip_bf16.h>

typedef __attribute__((ext_vector_type(16))) _Float16 v16h;
typedef __attribute__((ext_vector_type(8)))  float    v8f;
typedef __attribute__((ext_vector_type(4)))  int      v4i_t;

#define WSZ    8
#define SHIFT  4
#define HEADS  6
#define BATCH  2
#define IMG_H  256
#define IMG_W  256
#define CH     192
#define HID    768
#define RCH    24            // CH / 8
#define HD     32            // CH / HEADS
#define NTOK   64            // WSZ*WSZ
#define ROWS   (BATCH * IMG_H * IMG_W)   // 131072

// ---------------------------------------------------------------------------
// Async global->LDS staging (CDNA5 GLOBAL_LOAD_ASYNC_TO_LDS_B128, ASYNCcnt).
// Guarded by __has_builtin so the non-async toolchain still compiles.
// Builtin signature (probe-confirmed by diagnostic): (v4i_t __device__* src,
// v4i_t LDS* dst, imm offset, imm cpol) -- non-const pointers.
// ---------------------------------------------------------------------------
#if defined(__gfx1250__) && __has_builtin(__builtin_amdgcn_global_load_async_to_lds_b128) && __has_builtin(__builtin_amdgcn_s_wait_asynccnt)
#define ASYNC_LDS_COPY 1
typedef __attribute__((address_space(3))) v4i_t lds_v4i;
typedef __attribute__((address_space(1))) v4i_t glb_v4i;
#endif

// copy 96 bytes (6 x 16B) per thread from global to LDS
__device__ __forceinline__ void stage96(_Float16* dst, const _Float16* src) {
#if defined(ASYNC_LDS_COPY)
  lds_v4i* d = (lds_v4i*)dst;
  glb_v4i* s = (glb_v4i*)src;
#pragma unroll
  for (int u = 0; u < 6; ++u)
    __builtin_amdgcn_global_load_async_to_lds_b128(s + u, d + u, 0, 0);
#else
  const uint4* s = reinterpret_cast<const uint4*>(src);
  uint4* d = reinterpret_cast<uint4*>(dst);
#pragma unroll
  for (int u = 0; u < 6; ++u) d[u] = s[u];
#endif
}

__device__ __forceinline__ void stage_wait() {
#if defined(ASYNC_LDS_COPY)
  __builtin_amdgcn_s_wait_asynccnt(0);
#endif
}

// ---------------------------------------------------------------------------
// WMMA fragment helpers (wave32, v_wmma_f32_16x16x32_f16 layouts per CDNA5 ISA)
// ---------------------------------------------------------------------------

// A fragment: 16x32 f16, row-major source with stride ld (in halfs).
// lane m = lane&15, group = lane>>4; slots 0..7 -> K = grp*8 + h,
// slots 8..15 -> K = 16 + grp*8 + (h-8).  Two 16B loads per lane.
__device__ __forceinline__ v16h frag_a(const _Float16* base, int ld) {
  const int lane = threadIdx.x & 31;
  const _Float16* row = base + (lane & 15) * ld + ((lane >> 4) << 3);
  union { uint4 u[2]; v16h v; } r;
  r.u[0] = *reinterpret_cast<const uint4*>(row);
  r.u[1] = *reinterpret_cast<const uint4*>(row + 16);
  return r.v;
}

// B fragment: 32x16 (KxN) where B[k][n] = W[n*ld + k] (i.e. W^T of a
// row-major weight matrix). lane n = lane&15, slots 0..7 -> K = grp*8 + h,
// slots 8..15 -> K = 16 + grp*8 + (h-8).  Two 16B loads per lane.
__device__ __forceinline__ v16h frag_bT(const _Float16* base, int ld) {
  const int lane = threadIdx.x & 31;
  const _Float16* row = base + (lane & 15) * ld + ((lane >> 4) << 3);
  union { uint4 u[2]; v16h v; } r;
  r.u[0] = *reinterpret_cast<const uint4*>(row);
  r.u[1] = *reinterpret_cast<const uint4*>(row + 16);
  return r.v;
}

__device__ __forceinline__ v8f wmma16(v16h a, v16h b, v8f c) {
  return __builtin_amdgcn_wmma_f32_16x16x32_f16(false, a, false, b,
                                                (short)0, c, false, false);
}

__device__ __forceinline__ float wsum32(float v) {
#pragma unroll
  for (int o = 16; o > 0; o >>= 1) v += __shfl_xor(v, o, 32);
  return v;
}
__device__ __forceinline__ float wmax32(float v) {
#pragma unroll
  for (int o = 16; o > 0; o >>= 1) v = fmaxf(v, __shfl_xor(v, o, 32));
  return v;
}

// ---------------------------------------------------------------------------
// LayerNorm (one wave per 192-wide row) -> f16
// ---------------------------------------------------------------------------
__global__ void __launch_bounds__(256)
ln_f16_kernel(const float* __restrict__ x, const float* __restrict__ g,
              const float* __restrict__ b, _Float16* __restrict__ out) {
  const int wv = threadIdx.x >> 5, lane = threadIdx.x & 31;
  const int row = blockIdx.x * 8 + wv;
  const float* xr = x + (size_t)row * CH;
  float v[6]; float s = 0.f;
#pragma unroll
  for (int i = 0; i < 6; ++i) { v[i] = xr[lane + 32 * i]; s += v[i]; }
  s = wsum32(s);
  const float mean = s * (1.0f / CH);
  float q = 0.f;
#pragma unroll
  for (int i = 0; i < 6; ++i) { float d = v[i] - mean; q += d * d; }
  q = wsum32(q);
  const float inv = rsqrtf(q * (1.0f / CH) + 1e-5f);
  _Float16* orow = out + (size_t)row * CH;
#pragma unroll
  for (int i = 0; i < 6; ++i) {
    int c = lane + 32 * i;
    orow[c] = (_Float16)((v[i] - mean) * inv * g[c] + b[c]);
  }
}

__global__ void f32_to_f16_kernel(const float* __restrict__ s,
                                  _Float16* __restrict__ d, int n) {
  int i = blockIdx.x * 256 + threadIdx.x;
  if (i < n) d[i] = (_Float16)s[i];
}

__global__ void zero_kernel(float* p, int n) {
  int i = blockIdx.x * 256 + threadIdx.x;
  if (i < n) p[i] = 0.f;
}

// ---------------------------------------------------------------------------
// Fused shifted-window attention: one block per 8x8 window (2048 blocks).
// QKV GEMM + rel-pos bias + shift mask + softmax + P@V, all in LDS.
// ---------------------------------------------------------------------------
__device__ __forceinline__ int sregion(int p) {
  return (p < IMG_H - WSZ) ? 0 : ((p < IMG_H - SHIFT) ? 1 : 2);
}

__global__ void __launch_bounds__(256)
win_attn_kernel(const _Float16* __restrict__ xn, const _Float16* __restrict__ wqkv,
                const float* __restrict__ qkvb, const float* __restrict__ rpb,
                _Float16* __restrict__ aout) {
  __shared__ __align__(16) _Float16 Xs [NTOK * CH];    // 24.0 KB
  __shared__ __align__(16) _Float16 Qs [NTOK * HD];    //  4.0 KB
  __shared__ __align__(16) _Float16 Ksm[NTOK * HD];    //  4.0 KB
  __shared__ __align__(16) _Float16 Vt [HD * NTOK];    //  4.0 KB (d-major)
  __shared__ __align__(16) float    Ss [NTOK * NTOK];  // 16.0 KB
  __shared__ __align__(16) _Float16 Ps [NTOK * NTOK];  //  8.0 KB

  const int blk = blockIdx.x;
  const int bb = blk >> 10;
  const int w  = blk & 1023;
  const int wh = w >> 5, ww = w & 31;
  const int tid  = threadIdx.x;
  const int wv   = tid >> 5, lane = tid & 31;
  const int nsub = lane & 15, m0 = (lane >> 4) << 3;

  // gather shifted window tile: row n=(i,j) <- xn[(wh*8+i+4)%256][(ww*8+j+4)%256]
  {
    const int rowIdx = tid >> 2, quarter = tid & 3;
    const int i = rowIdx >> 3, j = rowIdx & 7;
    const int sh = (wh * WSZ + i + SHIFT) & (IMG_H - 1);
    const int sw = (ww * WSZ + j + SHIFT) & (IMG_W - 1);
    stage96(Xs + rowIdx * CH + quarter * 48,
            xn + (((size_t)bb * IMG_H + sh) * IMG_W + sw) * CH + quarter * 48);
    stage_wait();
  }
  __syncthreads();

  for (int h = 0; h < HEADS; ++h) {
    // ---- per-head QKV: 24 WMMA tiles (3 mats x 4 Mt x 2 Nt), K = 6 steps ----
#pragma unroll
    for (int it = 0; it < 3; ++it) {
      const int t = wv + it * 8;
      const int mat = t >> 3;
      const int mt = (t & 7) >> 1, nt = t & 1;
      const int oBase = mat * CH + h * HD + nt * 16;
      v8f c = {};
#pragma unroll
      for (int kk = 0; kk < 6; ++kk)
        c = wmma16(frag_a(Xs + mt * 16 * CH + kk * 32, CH),
                   frag_bT(wqkv + (size_t)oBase * CH + kk * 32, CH), c);
      const float bias = qkvb[oBase + nsub];
      if (mat == 0) {
#pragma unroll
        for (int r = 0; r < 8; ++r)
          Qs[(mt * 16 + m0 + r) * HD + nt * 16 + nsub] =
              (_Float16)((c[r] + bias) * 0.17677669529663687f);  // hd^-0.5
      } else if (mat == 1) {
#pragma unroll
        for (int r = 0; r < 8; ++r)
          Ksm[(mt * 16 + m0 + r) * HD + nt * 16 + nsub] = (_Float16)(c[r] + bias);
      } else {
#pragma unroll
        for (int r = 0; r < 8; ++r)
          Vt[(nt * 16 + nsub) * NTOK + mt * 16 + m0 + r] = (_Float16)(c[r] + bias);
      }
    }
    __syncthreads();

    // ---- scores S = Q @ K^T (+bias +mask): 16 tiles, shared Q fragment ----
    {
      const int mt = wv >> 1, ntg = (wv & 1) * 2;
      const v16h aq = frag_a(Qs + mt * 16 * HD, HD);   // one K=32 step
#pragma unroll
      for (int j = 0; j < 2; ++j) {
        const int nt = ntg + j;
        v8f c = {};
        c = wmma16(aq, frag_bT(Ksm + nt * 16 * HD, HD), c);
        const int ncol = nt * 16 + nsub;
        const int i2 = ncol >> 3, j2 = ncol & 7;
        const int regN = sregion(wh * WSZ + i2) * 3 + sregion(ww * WSZ + j2);
#pragma unroll
        for (int r = 0; r < 8; ++r) {
          const int mrow = mt * 16 + m0 + r;
          const int i1 = mrow >> 3, j1 = mrow & 7;
          const int regM = sregion(wh * WSZ + i1) * 3 + sregion(ww * WSZ + j1);
          const int ridx = (i1 - i2 + 7) * 15 + (j1 - j2 + 7);
          float val = c[r] + rpb[ridx * HEADS + h];
          if (regM != regN) val -= 100.0f;
          Ss[mrow * NTOK + ncol] = val;
        }
      }
    }
    __syncthreads();

    // ---- softmax over rows of 64 (8 rows per wave, 2 elems per lane) ----
#pragma unroll
    for (int rr = 0; rr < 8; ++rr) {
      const int row = wv * 8 + rr;
      const float v0 = Ss[row * NTOK + lane], v1 = Ss[row * NTOK + 32 + lane];
      const float mx = wmax32(fmaxf(v0, v1));
      const float e0 = __expf(v0 - mx), e1 = __expf(v1 - mx);
      const float inv = 1.0f / wsum32(e0 + e1);
      Ps[row * NTOK + lane]      = (_Float16)(e0 * inv);
      Ps[row * NTOK + 32 + lane] = (_Float16)(e1 * inv);
    }
    __syncthreads();

    // ---- O = P @ V: 8 tiles (one per wave), K = 2 steps; scatter un-shifted ----
    {
      const int mt = wv >> 1, nt = wv & 1;
      v8f c = {};
#pragma unroll
      for (int kk = 0; kk < 2; ++kk)
        c = wmma16(frag_a(Ps + mt * 16 * NTOK + kk * 32, NTOK),
                   frag_bT(Vt + nt * 16 * NTOK + kk * 32, NTOK), c);
      const int ch = h * HD + nt * 16 + nsub;
#pragma unroll
      for (int r = 0; r < 8; ++r) {
        const int mrow = mt * 16 + m0 + r;
        const int i1 = mrow >> 3, j1 = mrow & 7;
        const int oh = (wh * WSZ + i1 + SHIFT) & (IMG_H - 1);  // reverse roll
        const int ow = (ww * WSZ + j1 + SHIFT) & (IMG_W - 1);
        aout[(((size_t)bb * IMG_H + oh) * IMG_W + ow) * CH + ch] = (_Float16)c[r];
      }
    }
    __syncthreads();
  }
}

// ---------------------------------------------------------------------------
// proj GEMM + residual: x1 = attn @ proj_w^T + b + x  (64 rows x 192 cols/block)
// ---------------------------------------------------------------------------
__global__ void __launch_bounds__(256)
proj_res_kernel(const _Float16* __restrict__ a, const _Float16* __restrict__ wp,
                const float* __restrict__ pb, const float* __restrict__ shortcut,
                float* __restrict__ x1) {
  __shared__ __align__(16) _Float16 As[64 * CH];
  const int row0 = blockIdx.x * 64;
  const int tid = threadIdx.x, wv = tid >> 5, lane = tid & 31;
  const int nsub = lane & 15, m0 = (lane >> 4) << 3;
  {
    const int rowIdx = tid >> 2, quarter = tid & 3;
    stage96(As + rowIdx * CH + quarter * 48,
            a + (size_t)(row0 + rowIdx) * CH + quarter * 48);
    stage_wait();
  }
  __syncthreads();
  // wave -> (mt, 6-wide nt group): A fragments reused across 6 N-tiles
  const int mt = wv >> 1, ntg = (wv & 1) * 6;
  v8f acc[6] = {};
#pragma unroll
  for (int kk = 0; kk < 6; ++kk) {
    const v16h av = frag_a(As + mt * 16 * CH + kk * 32, CH);
#pragma unroll
    for (int j = 0; j < 6; ++j)
      acc[j] = wmma16(av, frag_bT(wp + (size_t)((ntg + j) * 16) * CH + kk * 32, CH), acc[j]);
  }
#pragma unroll
  for (int j = 0; j < 6; ++j) {
    const int col = (ntg + j) * 16 + nsub;
    const float bias = pb[col];
#pragma unroll
    for (int r = 0; r < 8; ++r) {
      const size_t row = (size_t)row0 + mt * 16 + m0 + r;
      x1[row * CH + col] = acc[j][r] + bias + shortcut[row * CH + col];
    }
  }
}

// ---------------------------------------------------------------------------
// fc1 GEMM + exact GELU -> f16   (grid: rows/64 x 4 column slabs of 192)
// ---------------------------------------------------------------------------
__global__ void __launch_bounds__(256)
fc1_gelu_kernel(const _Float16* __restrict__ a, const _Float16* __restrict__ w1,
                const float* __restrict__ b1, _Float16* __restrict__ out) {
  __shared__ __align__(16) _Float16 As[64 * CH];
  const int row0 = blockIdx.x * 64;
  const int col0 = blockIdx.y * 192;
  const int tid = threadIdx.x, wv = tid >> 5, lane = tid & 31;
  const int nsub = lane & 15, m0 = (lane >> 4) << 3;
  {
    const int rowIdx = tid >> 2, quarter = tid & 3;
    stage96(As + rowIdx * CH + quarter * 48,
            a + (size_t)(row0 + rowIdx) * CH + quarter * 48);
    stage_wait();
  }
  __syncthreads();
  const int mt = wv >> 1, ntg = (wv & 1) * 6;
  v8f acc[6] = {};
#pragma unroll
  for (int kk = 0; kk < 6; ++kk) {
    const v16h av = frag_a(As + mt * 16 * CH + kk * 32, CH);
#pragma unroll
    for (int j = 0; j < 6; ++j)
      acc[j] = wmma16(av, frag_bT(w1 + (size_t)(col0 + (ntg + j) * 16) * CH + kk * 32, CH), acc[j]);
  }
#pragma unroll
  for (int j = 0; j < 6; ++j) {
    const int col = col0 + (ntg + j) * 16 + nsub;
    const float bias = b1[col];
#pragma unroll
    for (int r = 0; r < 8; ++r) {
      const size_t row = (size_t)row0 + mt * 16 + m0 + r;
      const float v = acc[j][r] + bias;
      out[row * HID + col] = (_Float16)(0.5f * v * (1.0f + erff(v * 0.70710678118654752f)));
    }
  }
}

// ---------------------------------------------------------------------------
// fc2 GEMM + residual -> d_out (x2).  K=768 streamed from global; each wave
// keeps 6 accumulators so every A fragment is loaded once per k-step.
// ---------------------------------------------------------------------------
__global__ void __launch_bounds__(256)
fc2_res_kernel(const _Float16* __restrict__ hmid, const _Float16* __restrict__ w2,
               const float* __restrict__ b2, const float* __restrict__ x1,
               float* __restrict__ out) {
  const int row0 = blockIdx.x * 64;
  const int tid = threadIdx.x, wv = tid >> 5, lane = tid & 31;
  const int nsub = lane & 15, m0 = (lane >> 4) << 3;
  const int mt = wv >> 1, ntg = (wv & 1) * 6;
  const _Float16* abase = hmid + (size_t)(row0 + mt * 16) * HID;
  v8f acc[6] = {};
  for (int kk = 0; kk < 24; ++kk) {
    if (kk + 2 < 24)
      __builtin_prefetch(abase + (lane & 15) * HID + (kk + 2) * 32, 0, 1);
    const v16h av = frag_a(abase + kk * 32, HID);
#pragma unroll
    for (int j = 0; j < 6; ++j)
      acc[j] = wmma16(av, frag_bT(w2 + (size_t)((ntg + j) * 16) * HID + kk * 32, HID), acc[j]);
  }
#pragma unroll
  for (int j = 0; j < 6; ++j) {
    const int col = (ntg + j) * 16 + nsub;
    const float bias = b2[col];
#pragma unroll
    for (int r = 0; r < 8; ++r) {
      const size_t row = (size_t)row0 + mt * 16 + m0 + r;
      out[row * CH + col] = acc[j][r] + bias + x1[row * CH + col];
    }
  }
}

// ---------------------------------------------------------------------------
// LCE: conv1x1 (192->24), conv3x3 (24->24, SAME), conv1x1 (24->192)+LeakyReLU
// with SE channel gate, fused add into d_out.
// ---------------------------------------------------------------------------
__global__ void __launch_bounds__(256)
lce_conv1_kernel(const _Float16* __restrict__ xn, const float* __restrict__ w1,
                 const float* __restrict__ b1, float* __restrict__ y1) {
  __shared__ __align__(16) _Float16 Xs[64 * CH];
  __shared__ float W1s[RCH * CH];
  const int p0 = blockIdx.x * 64;
  const int tid = threadIdx.x;
  for (int i = tid; i < RCH * CH; i += 256) W1s[i] = w1[i];
  {
    const int rowIdx = tid >> 2, quarter = tid & 3;
    stage96(Xs + rowIdx * CH + quarter * 48,
            xn + (size_t)(p0 + rowIdx) * CH + quarter * 48);
    stage_wait();
  }
  __syncthreads();
#pragma unroll
  for (int i = 0; i < 6; ++i) {
    const int o = tid + 256 * i;          // 0..1535
    const int pix = o & 63, ch = o >> 6;  // 64 pixels x 24 channels
    float s = b1[ch];
    for (int c = 0; c < CH; ++c) s += (float)Xs[pix * CH + c] * W1s[ch * CH + c];
    const int p = p0 + pix;
    const int b = p >> 16, hw = p & 65535;
    y1[((size_t)b * RCH + ch) * 65536 + hw] = s;
  }
}

__global__ void __launch_bounds__(256)
lce_conv2_kernel(const float* __restrict__ y1, const float* __restrict__ w2,
                 const float* __restrict__ b2, float* __restrict__ y2) {
  __shared__ float W2s[RCH * RCH * 9];
  const int tid = threadIdx.x;
  for (int i = tid; i < RCH * RCH * 9; i += 256) W2s[i] = w2[i];
  __syncthreads();
  const int p = blockIdx.x * 256 + tid;
  const int b = p >> 16, hw = p & 65535;
  const int hh = hw >> 8, ww = hw & 255;
  float acc[RCH];
#pragma unroll
  for (int oc = 0; oc < RCH; ++oc) acc[oc] = b2[oc];
  for (int ic = 0; ic < RCH; ++ic) {
    const float* plane = y1 + ((size_t)b * RCH + ic) * 65536;
    float nb[9];
#pragma unroll
    for (int dy = -1; dy <= 1; ++dy)
#pragma unroll
      for (int dx = -1; dx <= 1; ++dx) {
        const int yy = hh + dy, xx = ww + dx;
        float v = 0.f;
        if (yy >= 0 && yy < IMG_H && xx >= 0 && xx < IMG_W) v = plane[yy * IMG_W + xx];
        nb[(dy + 1) * 3 + dx + 1] = v;
      }
#pragma unroll
    for (int oc = 0; oc < RCH; ++oc) {
      const float* wk = &W2s[(oc * RCH + ic) * 9];
      float s = 0.f;
#pragma unroll
      for (int t = 0; t < 9; ++t) s += nb[t] * wk[t];
      acc[oc] += s;
    }
  }
#pragma unroll
  for (int oc = 0; oc < RCH; ++oc)
    y2[((size_t)b * RCH + oc) * 65536 + hw] = acc[oc];
}

__global__ void __launch_bounds__(256)
lce_conv3_kernel(const float* __restrict__ y2, const float* __restrict__ w3,
                 const float* __restrict__ b3, _Float16* __restrict__ y16,
                 float* __restrict__ csum) {
  __shared__ float W3s[CH * RCH];
  const int tid = threadIdx.x, lane = tid & 31;
  for (int i = tid; i < CH * RCH; i += 256) W3s[i] = w3[i];
  __syncthreads();
  const int p = blockIdx.x * 256 + tid;
  const int b = p >> 16, hw = p & 65535;
  float z[RCH];
#pragma unroll
  for (int rr = 0; rr < RCH; ++rr) z[rr] = y2[((size_t)b * RCH + rr) * 65536 + hw];
  for (int c = 0; c < CH; ++c) {
    float s = b3[c];
#pragma unroll
    for (int rr = 0; rr < RCH; ++rr) s += z[rr] * W3s[c * RCH + rr];
    s = (s >= 0.f) ? s : 0.2f * s;                 // LeakyReLU(0.2)
    y16[((size_t)b * CH + c) * 65536 + hw] = (_Float16)s;
    const float pw = wsum32(s);                    // SE pool partial
    if (lane == 0) atomicAdd(&csum[b * CH + c], pw);
  }
}

__global__ void lce_gate_kernel(const float* __restrict__ csum,
                                const float* __restrict__ f1,
                                const float* __restrict__ f2,
                                float* __restrict__ sg) {
  __shared__ float mean[BATCH * CH];
  __shared__ float t1[BATCH * RCH];
  const int tid = threadIdx.x;
  for (int i = tid; i < BATCH * CH; i += 256) mean[i] = csum[i] * (1.0f / 65536.0f);
  __syncthreads();
  if (tid < BATCH * RCH) {
    const int b = tid / RCH, rr = tid % RCH;
    float s = 0.f;
    for (int c = 0; c < CH; ++c) s += mean[b * CH + c] * f1[rr * CH + c];
    t1[tid] = fmaxf(s, 0.f);
  }
  __syncthreads();
  for (int i = tid; i < BATCH * CH; i += 256) {
    const int b = i / CH, c = i % CH;
    float s = 0.f;
#pragma unroll
    for (int rr = 0; rr < RCH; ++rr) s += t1[b * RCH + rr] * f2[c * RCH + rr];
    sg[i] = 1.0f / (1.0f + __expf(-s));
  }
}

__global__ void final_add_kernel(const _Float16* __restrict__ y16,
                                 const float* __restrict__ sg,
                                 float* __restrict__ out) {
  const size_t o = (size_t)blockIdx.x * 256 + threadIdx.x;
  const int b = (int)(o / ((size_t)65536 * CH));
  const size_t rem = o % ((size_t)65536 * CH);
  const int hw = (int)(rem / CH), c = (int)(rem % CH);
  out[o] += (float)y16[((size_t)b * CH + c) * 65536 + hw] * sg[b * CH + c];
}

// ---------------------------------------------------------------------------
extern "C" void kernel_launch(void* const* d_in, const int* in_sizes, int n_in,
                              void* d_out, int out_size, void* d_ws, size_t ws_size,
                              hipStream_t stream) {
  const float* x      = (const float*)d_in[0];
  const float* n1g    = (const float*)d_in[1];
  const float* n1b    = (const float*)d_in[2];
  const float* qkv_w  = (const float*)d_in[3];
  const float* qkv_b  = (const float*)d_in[4];
  const float* rpb    = (const float*)d_in[5];
  const float* proj_w = (const float*)d_in[6];
  const float* proj_b = (const float*)d_in[7];
  const float* n2g    = (const float*)d_in[8];
  const float* n2b    = (const float*)d_in[9];
  const float* fc1_w  = (const float*)d_in[10];
  const float* fc1_b  = (const float*)d_in[11];
  const float* fc2_w  = (const float*)d_in[12];
  const float* fc2_b  = (const float*)d_in[13];
  const float* c1w    = (const float*)d_in[14];
  const float* c1b    = (const float*)d_in[15];
  const float* c2w    = (const float*)d_in[16];
  const float* c2b    = (const float*)d_in[17];
  const float* c3w    = (const float*)d_in[18];
  const float* c3b    = (const float*)d_in[19];
  const float* lf1    = (const float*)d_in[20];
  const float* lf2    = (const float*)d_in[21];
  float* out = (float*)d_out;
  (void)in_sizes; (void)n_in; (void)out_size; (void)ws_size;

  char* ws = (char*)d_ws;
  size_t off = 0;
  auto salloc = [&](size_t bytes) -> void* {
    void* p = ws + off;
    off = (off + bytes + 255) & ~(size_t)255;
    return p;
  };
  _Float16* xn16    = (_Float16*)salloc((size_t)ROWS * CH * 2);
  _Float16* attn16  = (_Float16*)salloc((size_t)ROWS * CH * 2);
  float*    x1      = (float*)   salloc((size_t)ROWS * CH * 4);
  _Float16* h16     = (_Float16*)salloc((size_t)ROWS * CH * 2);
  _Float16* hmid    = (_Float16*)salloc((size_t)ROWS * HID * 2);
  _Float16* wqkv16  = (_Float16*)salloc((size_t)3 * CH * CH * 2);
  _Float16* wproj16 = (_Float16*)salloc((size_t)CH * CH * 2);
  _Float16* wfc1    = (_Float16*)salloc((size_t)HID * CH * 2);
  _Float16* wfc2    = (_Float16*)salloc((size_t)CH * HID * 2);
  float*    y1      = (float*)   salloc((size_t)BATCH * RCH * 65536 * 4);
  float*    y2      = (float*)   salloc((size_t)BATCH * RCH * 65536 * 4);
  _Float16* y16     = (_Float16*)salloc((size_t)BATCH * CH * 65536 * 2);
  float*    csum    = (float*)   salloc((size_t)BATCH * CH * 4);
  float*    sg      = (float*)   salloc((size_t)BATCH * CH * 4);

  zero_kernel<<<2, 256, 0, stream>>>(csum, BATCH * CH);
  f32_to_f16_kernel<<<(3 * CH * CH + 255) / 256, 256, 0, stream>>>(qkv_w, wqkv16, 3 * CH * CH);
  f32_to_f16_kernel<<<(CH * CH + 255) / 256, 256, 0, stream>>>(proj_w, wproj16, CH * CH);
  f32_to_f16_kernel<<<(HID * CH + 255) / 256, 256, 0, stream>>>(fc1_w, wfc1, HID * CH);
  f32_to_f16_kernel<<<(CH * HID + 255) / 256, 256, 0, stream>>>(fc2_w, wfc2, CH * HID);

  ln_f16_kernel<<<ROWS / 8, 256, 0, stream>>>(x, n1g, n1b, xn16);
  win_attn_kernel<<<BATCH * 1024, 256, 0, stream>>>(xn16, wqkv16, qkv_b, rpb, attn16);
  proj_res_kernel<<<ROWS / 64, 256, 0, stream>>>(attn16, wproj16, proj_b, x, x1);
  ln_f16_kernel<<<ROWS / 8, 256, 0, stream>>>(x1, n2g, n2b, h16);
  fc1_gelu_kernel<<<dim3(ROWS / 64, HID / 192), 256, 0, stream>>>(h16, wfc1, fc1_b, hmid);
  fc2_res_kernel<<<ROWS / 64, 256, 0, stream>>>(hmid, wfc2, fc2_b, x1, out);

  lce_conv1_kernel<<<ROWS / 64, 256, 0, stream>>>(xn16, c1w, c1b, y1);
  lce_conv2_kernel<<<ROWS / 256, 256, 0, stream>>>(y1, c2w, c2b, y2);
  lce_conv3_kernel<<<ROWS / 256, 256, 0, stream>>>(y2, c3w, c3b, y16, csum);
  lce_gate_kernel<<<1, 256, 0, stream>>>(csum, lf1, lf2, sg);
  final_add_kernel<<<(ROWS * CH) / 256, 256, 0, stream>>>(y16, sg, out);
}
